// traj_score_14791867368118
// MI455X (gfx1250) — matrix operand order repaired and checked by reading, not compiled, not run on hardware
//
#include <hip/hip_runtime.h>
#include <hip/hip_bf16.h>
#include <math.h>

typedef __attribute__((ext_vector_type(16))) _Float16 v16h;
typedef __attribute__((ext_vector_type(8)))  _Float16 v8h;
typedef __attribute__((ext_vector_type(8)))  float    v8f;

#define BQ 4096
#define NN 64
#define TT 30
#define XLEN 128
#define KSEL 6
#define HH 64
#define ALPHA 0.01f
#define THRESH 0.1f
#define PRE_LEN (NN * 2 * TT)   // 3840

__device__ __forceinline__ v16h ld16(const _Float16* p0, const _Float16* p1) {
    v8h lo = *(const v8h*)p0;
    v8h hi = *(const v8h*)p1;
    return __builtin_shufflevector(lo, hi, 0,1,2,3,4,5,6,7,8,9,10,11,12,13,14,15);
}

__device__ __forceinline__ float gelu_exact(float x) {
    return 0.5f * x * (1.0f + erff(x * 0.70710678118654752f));
}

// 64x64 = A(64x64 f16, row-major) @ B (given TRANSPOSED: Bt[n][k], f16), f32 out.
// 4 waves, wave w owns rows [16w,16w+16); per wave: 4 N-tiles x 2 K-steps of WMMA.
__device__ __forceinline__ void gemm64(const _Float16* __restrict__ A,
                                       const _Float16* __restrict__ Bt,
                                       float* __restrict__ C, int tid) {
    const int wave = tid >> 5;
    const int lane = tid & 31;
    const int half = lane >> 4;
    const int l    = lane & 15;
    const int mBase = wave * 16;

    v8f acc[4] = {};
    #pragma unroll
    for (int ks = 0; ks < 2; ++ks) {
        const int kb = ks * 32;
        // A fragment: lane group 0 -> K {kb..kb+7, kb+16..kb+23}; group 1 -> +8
        const _Float16* pa = A + (mBase + l) * 64 + kb + 8 * half;
        v16h af = ld16(pa, pa + 16);
        #pragma unroll
        for (int nt = 0; nt < 4; ++nt) {
            // B fragment: lane l = column; lane group 0 -> K {kb..kb+7, kb+16..kb+23}; group 1 -> +8
            const _Float16* pb = Bt + (nt * 16 + l) * 64 + kb + 8 * half;
            v16h bf = ld16(pb, pb + 16);
            acc[nt] = __builtin_amdgcn_wmma_f32_16x16x32_f16(
                false, af, false, bf, (short)0, acc[nt], false, false);
        }
    }
    #pragma unroll
    for (int nt = 0; nt < 4; ++nt) {
        #pragma unroll
        for (int r = 0; r < 8; ++r) {
            C[(mBase + 8 * half + r) * 64 + nt * 16 + l] = acc[nt][r];
        }
    }
}

__global__ __launch_bounds__(128)
void traj_score_main(const float* __restrict__ agent, const float* __restrict__ pre,
                     const float* __restrict__ labels, const float* __restrict__ vec,
                     const float* __restrict__ W1, const float* __restrict__ b1,
                     const float* __restrict__ g1, const float* __restrict__ be1,
                     const float* __restrict__ W2, const float* __restrict__ b2,
                     const float* __restrict__ g2, const float* __restrict__ be2,
                     const float* __restrict__ W3, const float* __restrict__ b3,
                     float* __restrict__ outTraj, float* __restrict__ outDis,
                     float* __restrict__ ws) {
    __shared__ float    preL[PRE_LEN];     // per-batch candidates (f32)
    __shared__ float    lab[2 * TT];
    __shared__ _Float16 Atile[64 * 64];
    __shared__ _Float16 Btile[64 * 64];    // transposed operand
    __shared__ float    Cmat[64 * 64];
    __shared__ float    svec[64];          // shared-feature contribution (+b1)
    __shared__ float    mu[64], rstd[64];
    __shared__ float    rowbuf[64];
    __shared__ float    logp[64];
    __shared__ float    red2[2];
    __shared__ int      idxA[64];
    __shared__ float    dvalA[64];
    __shared__ float    tdval[64];
    __shared__ int      topsIdx[KSEL];

    const int b   = blockIdx.x;
    const int tid = threadIdx.x;

    // ---- stage per-batch data ----
    const float* preB = pre + (size_t)b * PRE_LEN;
    for (int i = tid; i < PRE_LEN; i += 128) preL[i] = preB[i];
    for (int i = tid; i < 2 * TT; i += 128)  lab[i]  = labels[(size_t)b * 2 * TT + i];

    // shared part of layer1: s[j] = b1[j] + [vec,agent] @ W1[0:256, j]
    if (tid < 64) {
        float acc = b1[tid];
        const float* vb = vec   + (size_t)b * XLEN;
        const float* ab = agent + (size_t)b * XLEN;
        for (int k = 0; k < XLEN; ++k) acc += vb[k] * W1[k * HH + tid];
        for (int k = 0; k < XLEN; ++k) acc += ab[k] * W1[(XLEN + k) * HH + tid];
        svec[tid] = acc;
    }
    // Btile = W1[256:316,:]^T padded to 64 (Bt[n][k])
    for (int i = tid; i < 4096; i += 128) {
        int n = i >> 6, k = i & 63;
        Btile[n * 64 + k] = (k < 60) ? (_Float16)W1[(256 + k) * HH + n] : (_Float16)0.0f;
    }
    __syncthreads();
    // Atile = pre_b padded 64x64
    for (int i = tid; i < 4096; i += 128) {
        int n = i >> 6, k = i & 63;
        Atile[n * 64 + k] = (k < 60) ? (_Float16)preL[n * 60 + k] : (_Float16)0.0f;
    }
    __syncthreads();

    // ---- layer 1 GEMM + LN + GELU ----
    gemm64(Atile, Btile, Cmat, tid);
    __syncthreads();
    for (int i = tid; i < 4096; i += 128) Cmat[i] += svec[i & 63];
    __syncthreads();
    if (tid < 64) {
        const float* row = Cmat + tid * 64;
        float s = 0.f;
        for (int j = 0; j < 64; ++j) s += row[j];
        float m = s * (1.0f / 64.0f), v = 0.f;
        for (int j = 0; j < 64; ++j) { float d = row[j] - m; v += d * d; }
        mu[tid] = m; rstd[tid] = rsqrtf(v * (1.0f / 64.0f) + 1e-5f);
    }
    __syncthreads();
    for (int i = tid; i < 4096; i += 128) {
        int r = i >> 6, j = i & 63;
        float x = (Cmat[i] - mu[r]) * rstd[r] * g1[j] + be1[j];
        Atile[i] = (_Float16)gelu_exact(x);
    }
    for (int i = tid; i < 4096; i += 128) {        // Btile = W2^T
        int n = i >> 6, k = i & 63;
        Btile[n * 64 + k] = (_Float16)W2[k * HH + n];
    }
    __syncthreads();

    // ---- layer 2 GEMM + LN + GELU ----
    gemm64(Atile, Btile, Cmat, tid);
    __syncthreads();
    for (int i = tid; i < 4096; i += 128) Cmat[i] += b2[i & 63];
    __syncthreads();
    if (tid < 64) {
        const float* row = Cmat + tid * 64;
        float s = 0.f;
        for (int j = 0; j < 64; ++j) s += row[j];
        float m = s * (1.0f / 64.0f), v = 0.f;
        for (int j = 0; j < 64; ++j) { float d = row[j] - m; v += d * d; }
        mu[tid] = m; rstd[tid] = rsqrtf(v * (1.0f / 64.0f) + 1e-5f);
    }
    __syncthreads();
    for (int i = tid; i < 4096; i += 128) {
        int r = i >> 6, j = i & 63;
        float x = (Cmat[i] - mu[r]) * rstd[r] * g2[j] + be2[j];
        Cmat[i] = gelu_exact(x);
    }
    __syncthreads();

    // ---- logits + log_softmax over N=64 ----
    if (tid < 64) {
        float acc = b3[0];
        const float* row = Cmat + tid * 64;
        for (int j = 0; j < 64; ++j) acc += row[j] * W3[j];
        rowbuf[tid] = acc;
    }
    __syncthreads();
    if (tid == 0) {
        float mx = -INFINITY;
        for (int i = 0; i < 64; ++i) mx = fmaxf(mx, rowbuf[i]);
        float s = 0.f;
        for (int i = 0; i < 64; ++i) s += expf(rowbuf[i] - mx);
        red2[0] = mx; red2[1] = logf(s);
    }
    __syncthreads();
    if (tid < 64) logp[tid] = rowbuf[tid] - red2[0] - red2[1];

    // ---- dis = softmax(-dmax/alpha), L3 partial ----
    if (tid < 64) {
        float dm = -INFINITY;
        for (int t = 0; t < TT; ++t) {
            float dx = preL[tid * 60 + 2 * t]     - lab[2 * t];
            float dy = preL[tid * 60 + 2 * t + 1] - lab[2 * t + 1];
            dm = fmaxf(dm, dx * dx + dy * dy);
        }
        rowbuf[tid] = -dm / ALPHA;
    }
    __syncthreads();
    if (tid == 0) {
        float mx = -INFINITY;
        for (int i = 0; i < 64; ++i) mx = fmaxf(mx, rowbuf[i]);
        float s = 0.f;
        for (int i = 0; i < 64; ++i) s += expf(rowbuf[i] - mx);
        red2[0] = mx; red2[1] = s;
    }
    __syncthreads();
    if (tid < 64) {
        // mimic reference exp-then-log (keeps underflow/NaN semantics identical)
        float dis = expf(rowbuf[tid] - red2[0]) / red2[1];
        tdval[tid] = dis * (logf(dis) - logp[tid]);
    }
    __syncthreads();
    if (tid == 0) {
        float s = 0.f;
        for (int i = 0; i < 64; ++i) s += tdval[i];
        ws[b] = s;  // reduced (and /B) by second kernel
    }

    // ---- initial stable descending sort by logp ----
    if (tid < 64) {
        float v = logp[tid];
        int rank = 0;
        for (int i = 0; i < 64; ++i) {
            float vi = logp[i];
            rank += (vi > v) || (vi == v && i < tid);
        }
        idxA[rank] = tid; dvalA[rank] = v;
    }
    __syncthreads();

    // ---- greedy top-K selection ----
    for (int k = 0; k < KSEL; ++k) {
        const int   hidx = idxA[0];
        const float hval = dvalA[0];
        if (tid == 0) {
            topsIdx[k] = hidx;
            outDis[(size_t)b * KSEL + k] = expf(hval);
        }
        if (tid < 60) {
            outTraj[((size_t)b * KSEL + k) * 60 + tid] = preL[hidx * 60 + tid];
        }
        __syncthreads();  // topsIdx[k] visible

        const int m = 63 - k;
        int myi = 0; float myd = 0.f;
        if (tid < m) {
            myi = idxA[tid + 1];
            myd = dvalA[tid + 1];
            // far check on first timestep point
            float px = preL[myi * 60 + 0], py = preL[myi * 60 + 1];
            if (sqrtf(px * px + py * py) > 5.0f) myd = -1.0f;
            // proximity: max_t mean_tops ||p_t - top_t||
            float dm = -INFINITY;
            for (int t = 0; t < TT; ++t) {
                float x = preL[myi * 60 + 2 * t], y = preL[myi * 60 + 2 * t + 1];
                float s = 0.f;
                for (int kk = 0; kk <= k; ++kk) {
                    int ti = topsIdx[kk];
                    float dx = x - preL[ti * 60 + 2 * t];
                    float dy = y - preL[ti * 60 + 2 * t + 1];
                    s += sqrtf(dx * dx + dy * dy);
                }
                s /= (float)(k + 1);
                dm = fmaxf(dm, s);
            }
            if (dm < THRESH) myd *= 1.1f;
            tdval[tid] = myd;
        }
        __syncthreads();
        if (tid < m) {
            int rank = 0;
            for (int i = 0; i < m; ++i) {
                float vi = tdval[i];
                rank += (vi > myd) || (vi == myd && i < tid);   // stable desc
            }
            idxA[rank] = myi; dvalA[rank] = myd;
        }
        __syncthreads();
    }
}

__global__ __launch_bounds__(256)
void traj_score_reduce(const float* __restrict__ ws, float* __restrict__ out) {
    __shared__ float s[256];
    int tid = threadIdx.x;
    float acc = 0.f;
    for (int i = tid; i < BQ; i += 256) acc += ws[i];
    s[tid] = acc;
    __syncthreads();
    for (int off = 128; off > 0; off >>= 1) {
        if (tid < off) s[tid] += s[tid + off];
        __syncthreads();
    }
    if (tid == 0) out[0] = s[0] / (float)BQ;
}

extern "C" void kernel_launch(void* const* d_in, const int* in_sizes, int n_in,
                              void* d_out, int out_size, void* d_ws, size_t ws_size,
                              hipStream_t stream) {
    (void)in_sizes; (void)n_in; (void)out_size; (void)ws_size;
    const float* agent  = (const float*)d_in[0];
    const float* pre    = (const float*)d_in[1];
    const float* labels = (const float*)d_in[2];
    const float* vec    = (const float*)d_in[3];
    const float* W1  = (const float*)d_in[4];
    const float* b1  = (const float*)d_in[5];
    const float* g1  = (const float*)d_in[6];
    const float* be1 = (const float*)d_in[7];
    const float* W2  = (const float*)d_in[8];
    const float* b2  = (const float*)d_in[9];
    const float* g2  = (const float*)d_in[10];
    const float* be2 = (const float*)d_in[11];
    const float* W3  = (const float*)d_in[12];
    const float* b3  = (const float*)d_in[13];

    float* out     = (float*)d_out;
    float* outTraj = out + 1;                               // (B, K, T, 2)
    float* outDis  = out + 1 + (size_t)BQ * KSEL * TT * 2;  // (B, K)
    float* ws      = (float*)d_ws;                          // BQ partials

    traj_score_main<<<BQ, 128, 0, stream>>>(agent, pre, labels, vec,
                                            W1, b1, g1, be1,
                                            W2, b2, g2, be2,
                                            W3, b3,
                                            outTraj, outDis, ws);
    traj_score_reduce<<<1, 256, 0, stream>>>(ws, out);
}
